// MyViT_4483945857119
// MI455X (gfx1250) — compile-verified, matrix operand order checked by reading, not run on hardware
//
#include <hip/hip_runtime.h>

// ---------------------------------------------------------------------------
// ViT-Base forward for MI455X (gfx1250, wave32, WMMA).
// All GEMMs run on v_wmma_f32_16x16x32_bf16 (f32 accum, bf16 operands).
// GEMM uses double-buffered LDS staging + global_prefetch of the tile after
// next; attention runs QK^T and PV on WMMA with LDS-staged K / V^T / probs.
// ---------------------------------------------------------------------------

typedef __attribute__((ext_vector_type(16))) __bf16 v16bf;
typedef __attribute__((ext_vector_type(8)))  __bf16 v8bf;
typedef __attribute__((ext_vector_type(8)))  float  v8f;

// Model constants
constexpr int kB   = 64;
constexpr int kS   = 197;          // 14*14 patches + cls
constexpr int kSP  = 224;          // padded seq (14*16 = 7*32)
constexpr int kD   = 768;
constexpr int kNH  = 12;
constexpr int kDH  = 64;
constexpr int kMLP = 3072;
constexpr int kOUT = 1000;
constexpr int kL   = 12;
constexpr int kM   = kB * kS;      // 12608 token rows
constexpr int kMP  = kB * 196;     // 12544 patch rows

// ---------------------------------------------------------------------------
// Fragment helpers. 16-bit A/B fragment layout (CDNA5 ISA 7.12.2):
//   lanes 0-15 : M=lane,    K chunks {0..7, 16..23}
//   lanes 16-31: M=lane-16, K chunks {8..15, 24..31}
// => per lane: two contiguous 8-half (16B) loads at offsets c0 and c0+16,
//    c0 = (lane<16) ? 0 : 8.
// ---------------------------------------------------------------------------
__device__ __forceinline__ v16bf ld_frag(const __bf16* p) {
  v8bf lo = *(const v8bf*)(p);
  v8bf hi = *(const v8bf*)(p + 16);
  v16bf r;
#pragma unroll
  for (int i = 0; i < 8; ++i) { r[i] = lo[i]; r[i + 8] = hi[i]; }
  return r;
}
__device__ __forceinline__ v16bf zero_frag() {
  v16bf r;
#pragma unroll
  for (int i = 0; i < 16; ++i) r[i] = (__bf16)0.0f;
  return r;
}

// ---------------------------------------------------------------------------
// Tiled bf16 WMMA GEMM: C[M,N] = A[M,K] (row-major bf16) * Bw[K,N] (row-major
// bf16) + bias, with fused epilogues.
//   EPI 0: f32 out = acc+bias
//   EPI 1: bf16 out
//   EPI 2: bf16 out, exact GELU
//   EPI 3: f32 in-place residual: Cf += acc+bias
//   EPI 4: patch remap: row' = row + row/196 + 1 (skip cls slots), f32 out
// Block tile 128x128, BK=32, 256 threads = 8 waves (4 along M x 2 along N),
// wave tile 32x64 = 2x4 accumulators. Double-buffered LDS (one barrier per
// K-step), register-staged global fetch of next tile overlaps the WMMAs,
// global_prefetch_b8 warms the tile after next.
// ---------------------------------------------------------------------------
#define BM 128
#define BN 128
#define BK 32
#define LDT 40   // BK + 8 halfs pad

template <int EPI>
__global__ __launch_bounds__(256) void gemm_bf16_wmma(
    const __bf16* __restrict__ A, const __bf16* __restrict__ Bw,
    const float* __restrict__ bias, float* __restrict__ Cf,
    __bf16* __restrict__ Cb, int M, int N, int K) {
  __shared__ __bf16 As[2][BM][LDT];
  __shared__ __bf16 Bs[2][BN][LDT];   // stored transposed: [n][k]

  const int tid  = threadIdx.x;
  const int bm   = blockIdx.y * BM;
  const int bn   = blockIdx.x * BN;
  const int wv   = tid >> 5;
  const int lane = tid & 31;
  const int lm   = lane & 15;
  const int lh   = lane >> 4;
  const int c0   = lh * 8;
  const int wm   = (wv & 3) * 32;
  const int wn   = (wv >> 2) * 64;

  // global->LDS assignments
  const int aRow = tid >> 1;            // 128 rows, 2 threads/row
  const int aCol = (tid & 1) * 16;
  const int bRow = tid >> 3;            // 32 k-rows, 8 threads/row
  const int bCol = (tid & 7) * 16;
  const int gmA  = bm + aRow;

  // register staging for global->LDS
  v8bf   alo, ahi;
  __bf16 brg[16];

  auto loadA = [&](int k0) {
    if (gmA < M) {
      const __bf16* src = A + (size_t)gmA * K + k0 + aCol;
      alo = *(const v8bf*)src;
      ahi = *(const v8bf*)(src + 8);
    } else {
#pragma unroll
      for (int i = 0; i < 8; ++i) { alo[i] = (__bf16)0.0f; ahi[i] = (__bf16)0.0f; }
    }
  };
  auto loadB = [&](int k0) {
    const int gk  = k0 + bRow;
    const int gn0 = bn + bCol;
    const __bf16* src = Bw + (size_t)gk * N + gn0;
    if (gn0 + 16 <= N) {
      v8bf lo = *(const v8bf*)src;
      v8bf hi = *(const v8bf*)(src + 8);
#pragma unroll
      for (int i = 0; i < 8; ++i) { brg[i] = lo[i]; brg[8 + i] = hi[i]; }
    } else {
#pragma unroll
      for (int i = 0; i < 16; ++i) {
        const int gn = gn0 + i;
        brg[i] = (gn < N) ? src[i] : (__bf16)0.0f;
      }
    }
  };
  auto stage = [&](int buf) {
    *(v8bf*)&As[buf][aRow][aCol]     = alo;
    *(v8bf*)&As[buf][aRow][aCol + 8] = ahi;
#pragma unroll
    for (int i = 0; i < 16; ++i) Bs[buf][bCol + i][bRow] = brg[i];
  };

  v8f acc[2][4];
#pragma unroll
  for (int tm = 0; tm < 2; ++tm)
#pragma unroll
    for (int tn = 0; tn < 4; ++tn)
#pragma unroll
      for (int r = 0; r < 8; ++r) acc[tm][tn][r] = 0.0f;

  // prologue: stage K-tile 0 into buffer 0
  loadA(0);
  loadB(0);
  stage(0);
  __syncthreads();

  int buf = 0;
  for (int k0 = 0; k0 < K; k0 += BK) {
    const int  kn   = k0 + BK;
    const bool more = kn < K;
    if (more) {
      // fetch next tile into registers; overlaps with WMMA below
      loadA(kn);
      loadB(kn);
      if (kn + BK < K) {  // warm GL2/WGP$ for the tile after next
        if (gmA < M)
          __builtin_prefetch(A + (size_t)gmA * K + kn + BK + aCol, 0, 1);
        __builtin_prefetch(Bw + (size_t)(kn + BK + bRow) * N + bn + bCol, 0, 1);
      }
    }

    v16bf af[2], bfr[4];
#pragma unroll
    for (int tm = 0; tm < 2; ++tm)
      af[tm] = ld_frag(&As[buf][wm + tm * 16 + lm][c0]);
#pragma unroll
    for (int tn = 0; tn < 4; ++tn)
      bfr[tn] = ld_frag(&Bs[buf][wn + tn * 16 + lm][c0]);
#pragma unroll
    for (int tm = 0; tm < 2; ++tm)
#pragma unroll
      for (int tn = 0; tn < 4; ++tn)
        acc[tm][tn] = __builtin_amdgcn_wmma_f32_16x16x32_bf16(
            false, af[tm], false, bfr[tn], (short)0, acc[tm][tn], false, false);

    if (more) {
      stage(buf ^ 1);     // write next tile into the other buffer
      __syncthreads();    // one barrier per K-step
      buf ^= 1;
    }
  }

  // ---- epilogue. C layout: VGPR r -> row tile + r + lh*8, col = lm.
#pragma unroll
  for (int tm = 0; tm < 2; ++tm)
#pragma unroll
    for (int tn = 0; tn < 4; ++tn)
#pragma unroll
      for (int r = 0; r < 8; ++r) {
        const int row = bm + wm + tm * 16 + lh * 8 + r;
        const int col = bn + wn + tn * 16 + lm;
        if (col >= N || row >= M) continue;
        float v = acc[tm][tn][r] + bias[col];
        if (EPI == 0) {
          Cf[(size_t)row * N + col] = v;
        } else if (EPI == 1) {
          Cb[(size_t)row * N + col] = (__bf16)v;
        } else if (EPI == 2) {
          v = 0.5f * v * (1.0f + erff(v * 0.70710678118654752f));
          Cb[(size_t)row * N + col] = (__bf16)v;
        } else if (EPI == 3) {
          const size_t o = (size_t)row * N + col;
          Cf[o] = Cf[o] + v;
        } else if (EPI == 4) {
          const int rr = row + row / 196 + 1;   // skip cls slot per image
          Cf[(size_t)rr * kD + col] = v;
        }
      }
}

// ---------------------------------------------------------------------------
// Attention: one workgroup (128 thr = 4 waves) per (batch, head).
// LDS: K [224][72] bf16, V^T [64][232] bf16, per-wave scores [16][224] f32,
// per-wave probs [16][232] bf16  (~149 KB -> 2 WGs / 320KB WGP).
// ---------------------------------------------------------------------------
__global__ __launch_bounds__(128) void attn_wmma(
    const __bf16* __restrict__ qkv, __bf16* __restrict__ ao) {
  const int b = blockIdx.x / kNH;
  const int h = blockIdx.x % kNH;

  extern __shared__ __align__(16) char smem_raw[];
  __bf16* Ks = (__bf16*)smem_raw;                        // [224][72]
  __bf16* VT = Ks + kSP * 72;                            // [64][232]
  float*  Sc = (float*)(VT + kDH * 232);                 // 4 x [16][224]
  __bf16* Ps = (__bf16*)(Sc + 4 * 16 * kSP);             // 4 x [16][232]

  const int tid  = threadIdx.x;
  const int wv   = tid >> 5;
  const int lane = tid & 31;
  const int lm   = lane & 15;
  const int lh   = lane >> 4;
  const int c0   = lh * 8;

  // ---- stage K rows (row-major over d) and V transposed ([d][k])
  for (int idx = tid; idx < kSP * kDH; idx += 128) {
    const int kr = idx / kDH, d = idx % kDH;
    __bf16 kv = (__bf16)0.0f, vv = (__bf16)0.0f;
    if (kr < kS) {
      const size_t base = ((size_t)(b * kS + kr)) * (3 * kD) + h * (3 * kDH);
      kv = qkv[base + kDH + d];        // K
      vv = qkv[base + 2 * kDH + d];    // V
    }
    Ks[kr * 72 + d]   = kv;
    VT[d * 232 + kr]  = vv;
  }
  __syncthreads();

  float* sc = Sc + wv * 16 * kSP;
  __bf16* pp = Ps + wv * 16 * 232;

  for (int qt = wv; qt < kSP / 16; qt += 4) {
    const int qrow = qt * 16 + lm;
    // Q fragments straight from global (A layout: 2x16B per lane)
    v16bf aq[2];
#pragma unroll
    for (int ks = 0; ks < 2; ++ks) {
      if (qrow < kS) {
        const __bf16* qb =
            qkv + ((size_t)(b * kS + qrow)) * (3 * kD) + h * (3 * kDH) + ks * 32 + c0;
        aq[ks] = ld_frag(qb);
      } else {
        aq[ks] = zero_frag();
      }
    }
    // ---- scores = Q K^T / sqrt(dh): 14 key tiles x 2 k-steps
    for (int nt = 0; nt < kSP / 16; ++nt) {
      v8f acc;
#pragma unroll
      for (int r = 0; r < 8; ++r) acc[r] = 0.0f;
#pragma unroll
      for (int ks = 0; ks < 2; ++ks) {
        v16bf bk = ld_frag(&Ks[(nt * 16 + lm) * 72 + ks * 32 + c0]);
        acc = __builtin_amdgcn_wmma_f32_16x16x32_bf16(
            false, aq[ks], false, bk, (short)0, acc, false, false);
      }
#pragma unroll
      for (int r = 0; r < 8; ++r)
        sc[(r + lh * 8) * kSP + nt * 16 + lm] = acc[r] * 0.125f;
    }
    // ---- softmax over valid keys [0,197); lanes lh=0/1 split even/odd cols
    float* rowp = sc + lm * kSP;
    float mx = -3.0e38f;
    for (int c = lh; c < kS; c += 2) mx = fmaxf(mx, rowp[c]);
    mx = fmaxf(mx, __shfl_xor(mx, 16));
    float sum = 0.0f;
    for (int c = lh; c < kS; c += 2) {
      const float e = expf(rowp[c] - mx);
      rowp[c] = e;
      sum += e;
    }
    sum += __shfl_xor(sum, 16);
    const float inv = 1.0f / sum;
    __bf16* pr = pp + lm * 232;
    for (int c = lh; c < kS; c += 2) pr[c] = (__bf16)(rowp[c] * inv);
    for (int c = kS + lh; c < kSP; c += 2) pr[c] = (__bf16)0.0f;  // pad
    // ---- O = P V : 4 d-tiles x 7 k-steps of 32
    for (int dt = 0; dt < 4; ++dt) {
      v8f acc;
#pragma unroll
      for (int r = 0; r < 8; ++r) acc[r] = 0.0f;
#pragma unroll
      for (int ks = 0; ks < kSP / 32; ++ks) {
        v16bf ap = ld_frag(&pp[lm * 232 + ks * 32 + c0]);
        v16bf bv = ld_frag(&VT[(dt * 16 + lm) * 232 + ks * 32 + c0]);
        acc = __builtin_amdgcn_wmma_f32_16x16x32_bf16(
            false, ap, false, bv, (short)0, acc, false, false);
      }
#pragma unroll
      for (int r = 0; r < 8; ++r) {
        const int row = qt * 16 + r + lh * 8;
        if (row < kS)
          ao[((size_t)(b * kS + row)) * kD + h * kDH + dt * 16 + lm] =
              (__bf16)acc[r];
      }
    }
  }
}

// ---------------------------------------------------------------------------
// Elementwise / reduction helpers
// ---------------------------------------------------------------------------
__global__ void k_patchify(const float* __restrict__ img,
                           __bf16* __restrict__ out) {
  const long long i = (long long)blockIdx.x * blockDim.x + threadIdx.x;
  if (i >= (long long)kMP * kD) return;
  const int m = (int)(i / kD), q = (int)(i % kD);
  const int b = m / 196, p = m % 196;
  const int py = p / 14, px = p % 14;
  const int c = q / 256, ph = (q / 16) % 16, pw = q % 16;
  const size_t src =
      (((size_t)b * 3 + c) * 224 + (py * 16 + ph)) * 224 + (px * 16 + pw);
  out[i] = (__bf16)img[src];
}

__global__ void k_convert(const float* __restrict__ in, __bf16* __restrict__ out,
                          long long n) {
  const long long i = (long long)blockIdx.x * blockDim.x + threadIdx.x;
  if (i < n) out[i] = (__bf16)in[i];
}

// write cls rows and add sinusoidal positional embedding to all of X
__global__ void k_finalize_embed(float* __restrict__ X,
                                 const float* __restrict__ cls_tok) {
  const long long i = (long long)blockIdx.x * blockDim.x + threadIdx.x;
  if (i >= (long long)kM * kD) return;
  const int row = (int)(i / kD), d = (int)(i % kD);
  const int s = row % kS;
  const float base = (s == 0) ? cls_tok[d] : X[i];
  const float de = (float)(d & ~1);
  const float ang = (float)s * powf(10000.0f, -de * (1.0f / (float)kD));
  const float pe = (d & 1) ? cosf(ang) : sinf(ang);
  X[i] = base + pe;
}

__global__ __launch_bounds__(256) void k_ln(const float* __restrict__ X,
                                            const float* __restrict__ g,
                                            const float* __restrict__ bta,
                                            __bf16* __restrict__ out) {
  const int row = blockIdx.x, t = threadIdx.x;
  const float* x = X + (size_t)row * kD;
  __shared__ float red[256];
  float a0 = x[t], a1 = x[t + 256], a2 = x[t + 512];
  red[t] = a0 + a1 + a2;
  __syncthreads();
  for (int o = 128; o > 0; o >>= 1) {
    if (t < o) red[t] += red[t + o];
    __syncthreads();
  }
  const float mean = red[0] * (1.0f / (float)kD);
  __syncthreads();
  const float v0 = a0 - mean, v1 = a1 - mean, v2 = a2 - mean;
  red[t] = v0 * v0 + v1 * v1 + v2 * v2;
  __syncthreads();
  for (int o = 128; o > 0; o >>= 1) {
    if (t < o) red[t] += red[t + o];
    __syncthreads();
  }
  const float rstd = rsqrtf(red[0] * (1.0f / (float)kD) + 1e-5f);
  __bf16* o = out + (size_t)row * kD;
  o[t]       = (__bf16)(v0 * rstd * g[t]       + bta[t]);
  o[t + 256] = (__bf16)(v1 * rstd * g[t + 256] + bta[t + 256]);
  o[t + 512] = (__bf16)(v2 * rstd * g[t + 512] + bta[t + 512]);
}

__global__ void k_gather_cls(const float* __restrict__ X,
                             __bf16* __restrict__ out) {
  const int i = blockIdx.x * blockDim.x + threadIdx.x;
  if (i >= kB * kD) return;
  const int b = i / kD, d = i % kD;
  out[i] = (__bf16)X[((size_t)b * kS) * kD + d];
}

__global__ __launch_bounds__(256) void k_softmax_out(float* __restrict__ out) {
  const int b = blockIdx.x, t = threadIdx.x;
  float* p = out + (size_t)b * kOUT;
  __shared__ float red[256];
  float mx = -3.0e38f;
  for (int c = t; c < kOUT; c += 256) mx = fmaxf(mx, p[c]);
  red[t] = mx;
  __syncthreads();
  for (int o = 128; o > 0; o >>= 1) {
    if (t < o) red[t] = fmaxf(red[t], red[t + o]);
    __syncthreads();
  }
  mx = red[0];
  __syncthreads();
  float s = 0.0f;
  for (int c = t; c < kOUT; c += 256) {
    const float e = expf(p[c] - mx);
    p[c] = e;
    s += e;
  }
  red[t] = s;
  __syncthreads();
  for (int o = 128; o > 0; o >>= 1) {
    if (t < o) red[t] += red[t + o];
    __syncthreads();
  }
  const float inv = 1.0f / red[0];
  __syncthreads();
  for (int c = t; c < kOUT; c += 256) p[c] *= inv;
}

// ---------------------------------------------------------------------------
// Host orchestration
// ---------------------------------------------------------------------------
extern "C" void kernel_launch(void* const* d_in, const int* in_sizes, int n_in,
                              void* d_out, int out_size, void* d_ws,
                              size_t ws_size, hipStream_t stream) {
  (void)in_sizes; (void)n_in; (void)out_size; (void)ws_size;

  const float* images  = (const float*)d_in[0];
  const float* lm_w    = (const float*)d_in[1];
  const float* lm_b    = (const float*)d_in[2];
  const float* cls_tok = (const float*)d_in[3];
  const float* ln1_g   = (const float*)d_in[4];
  const float* ln1_b   = (const float*)d_in[5];
  const float* qkv_w   = (const float*)d_in[6];
  const float* qkv_b   = (const float*)d_in[7];
  const float* out_w   = (const float*)d_in[8];
  const float* out_b   = (const float*)d_in[9];
  const float* ln2_g   = (const float*)d_in[10];
  const float* ln2_b   = (const float*)d_in[11];
  const float* mlp1_w  = (const float*)d_in[12];
  const float* mlp1_b  = (const float*)d_in[13];
  const float* mlp2_w  = (const float*)d_in[14];
  const float* mlp2_b  = (const float*)d_in[15];
  const float* head_w  = (const float*)d_in[16];
  const float* head_b  = (const float*)d_in[17];
  float* outp = (float*)d_out;

  // workspace carve (256B aligned)
  size_t off = 0;
  auto carve = [&](size_t bytes) {
    void* p = (char*)d_ws + off;
    off += (bytes + 255) & ~(size_t)255;
    return p;
  };
  float*  X    = (float*)carve((size_t)kM * kD * 4);        // master f32
  __bf16* XN   = (__bf16*)carve((size_t)kM * kD * 2);       // LN out / patches
  __bf16* QKV  = (__bf16*)carve((size_t)kM * 3 * kD * 2);
  __bf16* AO   = (__bf16*)carve((size_t)kM * kD * 2);       // attn out
  __bf16* H1   = (__bf16*)carve((size_t)kM * kMLP * 2);     // MLP hidden
  __bf16* WB   = (__bf16*)carve((size_t)kD * kMLP * 2);     // weight staging
  __bf16* CLS  = (__bf16*)carve((size_t)kB * kD * 2);

  auto cdiv = [](long long a, long long b) { return (int)((a + b - 1) / b); };
  const dim3 blk256(256);

  // ---- patch embedding
  k_patchify<<<cdiv((long long)kMP * kD, 256), blk256, 0, stream>>>(images, XN);
  k_convert<<<cdiv((long long)kD * kD, 256), blk256, 0, stream>>>(
      lm_w, WB, (long long)kD * kD);
  gemm_bf16_wmma<4><<<dim3(kD / BN, kMP / BM), blk256, 0, stream>>>(
      XN, WB, lm_b, X, nullptr, kMP, kD, kD);
  k_finalize_embed<<<cdiv((long long)kM * kD, 256), blk256, 0, stream>>>(
      X, cls_tok);

  const dim3 gM((kD + BN - 1) / BN, (kM + BM - 1) / BM);          // N=768
  const dim3 gQKV((3 * kD + BN - 1) / BN, (kM + BM - 1) / BM);    // N=2304
  const dim3 gMLP((kMLP + BN - 1) / BN, (kM + BM - 1) / BM);      // N=3072
  const size_t attn_lds = (size_t)kSP * 72 * 2 + (size_t)kDH * 232 * 2 +
                          (size_t)4 * 16 * kSP * 4 + (size_t)4 * 16 * 232 * 2;

  for (int l = 0; l < kL; ++l) {
    // LN1 -> XN, QKV GEMM
    k_ln<<<kM, blk256, 0, stream>>>(X, ln1_g + l * kD, ln1_b + l * kD, XN);
    k_convert<<<cdiv((long long)kD * 3 * kD, 256), blk256, 0, stream>>>(
        qkv_w + (size_t)l * kD * 3 * kD, WB, (long long)kD * 3 * kD);
    gemm_bf16_wmma<1><<<gQKV, blk256, 0, stream>>>(
        XN, WB, qkv_b + l * 3 * kD, nullptr, QKV, kM, 3 * kD, kD);
    // attention
    attn_wmma<<<dim3(kB * kNH), dim3(128), attn_lds, stream>>>(QKV, AO);
    // out projection + residual
    k_convert<<<cdiv((long long)kD * kD, 256), blk256, 0, stream>>>(
        out_w + (size_t)l * kD * kD, WB, (long long)kD * kD);
    gemm_bf16_wmma<3><<<gM, blk256, 0, stream>>>(
        AO, WB, out_b + l * kD, X, nullptr, kM, kD, kD);
    // LN2 -> XN, MLP1 (GELU), MLP2 + residual
    k_ln<<<kM, blk256, 0, stream>>>(X, ln2_g + l * kD, ln2_b + l * kD, XN);
    k_convert<<<cdiv((long long)kD * kMLP, 256), blk256, 0, stream>>>(
        mlp1_w + (size_t)l * kD * kMLP, WB, (long long)kD * kMLP);
    gemm_bf16_wmma<2><<<gMLP, blk256, 0, stream>>>(
        XN, WB, mlp1_b + l * kMLP, nullptr, H1, kM, kMLP, kD);
    k_convert<<<cdiv((long long)kMLP * kD, 256), blk256, 0, stream>>>(
        mlp2_w + (size_t)l * kMLP * kD, WB, (long long)kMLP * kD);
    gemm_bf16_wmma<3><<<gM, blk256, 0, stream>>>(
        H1, WB, mlp2_b + l * kD, X, nullptr, kM, kD, kMLP);
  }

  // ---- head: logits = cls @ head_w + head_b, then softmax (in d_out)
  k_gather_cls<<<cdiv(kB * kD, 256), blk256, 0, stream>>>(X, CLS);
  k_convert<<<cdiv((long long)kD * kOUT, 256), blk256, 0, stream>>>(
      head_w, WB, (long long)kD * kOUT);
  gemm_bf16_wmma<0><<<dim3((kOUT + BN - 1) / BN, 1), blk256, 0, stream>>>(
      CLS, WB, head_b, outp, nullptr, kB, kOUT, kD);
  k_softmax_out<<<kB, blk256, 0, stream>>>(outp);
}
